// AttentionHead_22076131902133
// MI455X (gfx1250) — matrix operand order repaired and checked by reading, not compiled, not run on hardware
//
#include <hip/hip_runtime.h>
#include <hip/hip_bf16.h>

// Problem constants (from reference)
#define BB 8
#define SS 1024
#define DM 512
#define DK 64
#define NITER 40
#define SPARS 0.05f

typedef __bf16 bf16_t;
typedef __attribute__((ext_vector_type(16))) __bf16 bf16x16;
typedef __attribute__((ext_vector_type(8)))  __bf16 bf16x8;
typedef __attribute__((ext_vector_type(8)))  float  f32x8;

// LDS geometry for the ISTA kernel (bank-rotating pads)
#define TT 32      // t-rows per workgroup
#define SP 1032    // alpha pitch (bf16)
#define KP 72      // Ks pitch (bf16)
#define QP 68      // Qs pitch (f32)
#define RP 72      // R pitch (bf16)

#define OFF_ALPHA 0
#define OFF_KS    66048                  // TT*SP*2
#define OFF_QS    (66048 + 147456)       // + SS*KP*2
#define OFF_R     (66048 + 147456 + 8704)
#define OFF_ETA   (66048 + 147456 + 8704 + 4608)
#define SMEM_BYTES (66048 + 147456 + 8704 + 4608 + 128)

__device__ __forceinline__ bf16x16 cat8(bf16x8 lo, bf16x8 hi) {
  return __builtin_shufflevector(lo, hi, 0,1,2,3,4,5,6,7,8,9,10,11,12,13,14,15);
}

__device__ __forceinline__ f32x8 wmma_bf16(bf16x16 a, bf16x16 b, f32x8 c) {
  // 8 args: (neg_a, A, neg_b, B, c_mod, C, reuse_a, reuse_b)
  return __builtin_amdgcn_wmma_f32_16x16x32_bf16(false, a, false, b, (short)0, c,
                                                 false, false);
}

// ---- async global -> LDS copy (16 bytes per lane), ASYNCcnt-tracked ----
#if __has_builtin(__builtin_amdgcn_global_load_async_to_lds_b128)
#define HAVE_ASYNC_LDS 1
#else
#define HAVE_ASYNC_LDS 0
#endif

#if HAVE_ASYNC_LDS
typedef int v4i __attribute__((vector_size(4 * sizeof(int))));
typedef __attribute__((address_space(1))) v4i gv4i;   // global
typedef __attribute__((address_space(3))) v4i lv4i;   // LDS
#endif

__device__ __forceinline__ void copy16_g2l(const bf16_t* g, bf16_t* l) {
#if HAVE_ASYNC_LDS
  __builtin_amdgcn_global_load_async_to_lds_b128(
      (gv4i*)(uintptr_t)g,
      (lv4i*)(unsigned int)(uintptr_t)l,
      0, 0);
#else
  *(bf16x8*)l = *(const bf16x8*)g;
#endif
}

__device__ __forceinline__ void wait_async0() {
#if HAVE_ASYNC_LDS
#if __has_builtin(__builtin_amdgcn_s_wait_asynccnt)
  __builtin_amdgcn_s_wait_asynccnt(0);
#else
  asm volatile("s_wait_asynccnt 0x0" ::: "memory");
#endif
#endif
}

// ---------------------------------------------------------------------------
// Kernel 1: projections.  Per thread: one (b,s,j) element of Q,K,V.
// Emits Qs(f32), Ks(f32), Ks(bf16 row-major), Ks^T(bf16), V^T(bf16).
// ---------------------------------------------------------------------------
__global__ __launch_bounds__(256)
void proj_kernel(const float* __restrict__ x,
                 const float* __restrict__ Wq, const float* __restrict__ bq,
                 const float* __restrict__ Wk, const float* __restrict__ bk,
                 const float* __restrict__ Wv, const float* __restrict__ bv,
                 float* __restrict__ Qs, float* __restrict__ Ksf,
                 bf16_t* __restrict__ Ksb, bf16_t* __restrict__ KsT,
                 bf16_t* __restrict__ VT)
{
  __shared__ float xr[4][DM];
  const int tid  = threadIdx.x;
  const int row4 = blockIdx.x * 4;                 // 4 (b,s) rows per block
  for (int i = tid; i < 4 * DM; i += 256) {
    int r = i >> 9, c = i & (DM - 1);
    xr[r][c] = x[(size_t)(row4 + r) * DM + c];
  }
  __syncthreads();

  const int r  = tid >> 6;                         // 0..3
  const int j  = tid & 63;                         // dk index
  const int bs = row4 + r;
  const int b  = bs >> 10, s = bs & (SS - 1);

  float q = bq[j], k = bk[j], v = bv[j];
  #pragma unroll 4
  for (int d = 0; d < DM; ++d) {
    float xv = xr[r][d];
    q = fmaf(xv, Wq[d * DK + j], q);               // lanes j coalesce per d
    k = fmaf(xv, Wk[d * DK + j], k);
    v = fmaf(xv, Wv[d * DK + j], v);
  }
  const float qs = q * 0.125f, ks = k * 0.125f;    // /sqrt(64)
  const size_t o = (size_t)bs * DK + j;
  Qs[o]  = qs;
  Ksf[o] = ks;
  Ksb[o] = (bf16_t)ks;
  KsT[((size_t)b * DK + j) * SS + s] = (bf16_t)ks;
  VT [((size_t)b * DK + j) * SS + s] = (bf16_t)v;
}

// ---------------------------------------------------------------------------
// Kernel 2: eta[b,s] = 0.9 / (lam_max(Ks[b,:s+1]^T Ks[b,:s+1]) + 1e-8)
// Power iteration without forming G:  w = K^T (K v),  lam -> ||w||.
// ---------------------------------------------------------------------------
__global__ __launch_bounds__(256)
void eta_kernel(const float* __restrict__ Ksf, float* __restrict__ eta)
{
  __shared__ float u[SS];
  __shared__ float vv[DK];
  __shared__ float wpart[4][DK];
  __shared__ float ww[DK];
  __shared__ float nrm;
  const int bsid = blockIdx.x;
  const int b = bsid >> 10, s = bsid & (SS - 1);
  const int n = s + 1;
  const int tid = threadIdx.x;
  const int g   = tid >> 6;                        // 0..3 t-strip
  const int j   = tid & 63;                        // dk lane
  const float* K = Ksf + (size_t)b * SS * DK;

  if (tid < DK) vv[tid] = 0.125f;                  // 1/sqrt(64)
  __syncthreads();

  float lam = 0.f;
  for (int it = 0; it < 12; ++it) {
    for (int t = tid; t < n; t += 256) {           // u = K v
      const float* kr = K + (size_t)t * DK;
      float acc = 0.f;
      #pragma unroll
      for (int d = 0; d < DK; ++d) acc = fmaf(kr[d], vv[d], acc);
      u[t] = acc;
    }
    __syncthreads();
    {                                              // w = K^T u (all 256 threads)
      float acc = 0.f;
      for (int t = g; t < n; t += 4) acc = fmaf(K[(size_t)t * DK + j], u[t], acc);
      wpart[g][j] = acc;
    }
    __syncthreads();
    if (tid < DK)
      ww[tid] = (wpart[0][tid] + wpart[1][tid]) + (wpart[2][tid] + wpart[3][tid]);
    __syncthreads();
    if (tid == 0) {
      float sum = 0.f;
      #pragma unroll
      for (int d = 0; d < DK; ++d) sum = fmaf(ww[d], ww[d], sum);
      nrm = sqrtf(sum);
    }
    __syncthreads();
    lam = nrm;
    const float inv = (nrm > 0.f) ? 1.f / nrm : 0.f;
    if (tid < DK) vv[tid] = ww[tid] * inv;
    __syncthreads();
  }
  if (tid == 0) eta[(size_t)bsid] = 0.9f / (lam + 1e-8f);
}

// ---------------------------------------------------------------------------
// Kernel 3: ISTA.  One workgroup (8 waves) owns a 32-row t-tile of alpha,
// resident in LDS for all 40 iterations.  All GEMMs via v_wmma_f32_16x16x32_bf16.
// Dual accumulators break the WMMA->WMMA RAW hazard chain.
// ---------------------------------------------------------------------------
__global__ __launch_bounds__(256)
void ista_kernel(const float* __restrict__ Qs, const bf16_t* __restrict__ Ksb,
                 const bf16_t* __restrict__ KsT, const bf16_t* __restrict__ VT,
                 const float* __restrict__ eta, float* __restrict__ out)
{
  extern __shared__ char smem[];
  bf16_t* sAlpha = (bf16_t*)(smem + OFF_ALPHA);    // TT x SP
  bf16_t* sKs    = (bf16_t*)(smem + OFF_KS);       // SS x KP (row-major Ks[b])
  float*  sQs    = (float*) (smem + OFF_QS);       // TT x QP
  bf16_t* sR     = (bf16_t*)(smem + OFF_R);        // TT x RP  (Ka - Qs, bf16)
  float*  sEta   = (float*) (smem + OFF_ETA);      // TT

  const int b   = blockIdx.x >> 5;                 // 32 t-blocks per batch
  const int t0  = (blockIdx.x & 31) * TT;
  const int tid = threadIdx.x;
  const int lane = tid & 31;
  const int wave = tid >> 5;

  // ---- init: alpha = 0 ----
  for (int i = tid; i < (TT * SP) / 2; i += 256) ((unsigned int*)sAlpha)[i] = 0u;
  // ---- Ks[b] -> LDS (row-major, padded pitch), async when available ----
  const bf16_t* gK = Ksb + (size_t)b * SS * DK;
  for (int i = tid; i < SS * 8; i += 256) {
    int r = i >> 3, c = (i & 7) * 8;
    copy16_g2l(gK + r * DK + c, sKs + r * KP + c);
  }
  // ---- Qs tile + eta ----
  const float* gQ = Qs + ((size_t)b * SS + t0) * DK;
  for (int i = tid; i < TT * DK; i += 256) {
    int r = i >> 6, c = i & 63;
    sQs[r * QP + c] = gQ[r * DK + c];
  }
  if (tid < TT) sEta[tid] = eta[(size_t)b * SS + t0 + tid];
  wait_async0();
  __syncthreads();

  // per-wave step-1 / output tile coords
  const int tr   = wave >> 2;                      // 0..1  (row block of 16)
  const int nc   = wave & 3;                       // 0..3  (dk col block of 16)
  const int mrow = lane & 15;
  const int half = (lane < 16) ? 0 : 8;            // unused (A uses bko too)
  const int bko  = (lane < 16) ? 0 : 16;           // B-operand k sub-offset
  const int arow = tr * 16 + mrow;                 // A row within t-tile
  const int bn1  = nc * 16 + mrow;                 // B column (dk index)
  const bf16_t* gKT = KsT + (size_t)b * DK * SS;

  for (int it = 0; it < NITER; ++it) {
    // ---- Step 1: R = alpha @ Ks - Qs  (32x64, one 16x16 tile per wave) ----
    f32x8 acc0 = {}, acc1 = {};
    const int nch = (t0 + tr * 16 + 16 + 31) >> 5; // causal k bound
    int kc = 0;
    for (; kc + 1 < nch; kc += 2) {
      {
        const int kb = kc * 32;
        __builtin_prefetch(gKT + (size_t)bn1 * SS + kb + 128, 0, 3);
        bf16x8 alo = *(const bf16x8*)(sAlpha + arow * SP + kb + bko);
        bf16x8 ahi = *(const bf16x8*)(sAlpha + arow * SP + kb + bko + 8);
        bf16x8 blo = *(const bf16x8*)(gKT + (size_t)bn1 * SS + kb + bko);
        bf16x8 bhi = *(const bf16x8*)(gKT + (size_t)bn1 * SS + kb + bko + 8);
        acc0 = wmma_bf16(cat8(alo, ahi), cat8(blo, bhi), acc0);
      }
      {
        const int kb = kc * 32 + 32;
        bf16x8 alo = *(const bf16x8*)(sAlpha + arow * SP + kb + bko);
        bf16x8 ahi = *(const bf16x8*)(sAlpha + arow * SP + kb + bko + 8);
        bf16x8 blo = *(const bf16x8*)(gKT + (size_t)bn1 * SS + kb + bko);
        bf16x8 bhi = *(const bf16x8*)(gKT + (size_t)bn1 * SS + kb + bko + 8);
        acc1 = wmma_bf16(cat8(alo, ahi), cat8(blo, bhi), acc1);
      }
    }
    if (kc < nch) {
      const int kb = kc * 32;
      bf16x8 alo = *(const bf16x8*)(sAlpha + arow * SP + kb + bko);
      bf16x8 ahi = *(const bf16x8*)(sAlpha + arow * SP + kb + bko + 8);
      bf16x8 blo = *(const bf16x8*)(gKT + (size_t)bn1 * SS + kb + bko);
      bf16x8 bhi = *(const bf16x8*)(gKT + (size_t)bn1 * SS + kb + bko + 8);
      acc0 = wmma_bf16(cat8(alo, ahi), cat8(blo, bhi), acc0);
    }
    {
      const int n = nc * 16 + mrow;
      #pragma unroll
      for (int i = 0; i < 8; ++i) {
        const int m = tr * 16 + i + ((lane < 16) ? 0 : 8);
        const float r = (acc0[i] + acc1[i]) - sQs[m * QP + n];
        sR[m * RP + n] = (bf16_t)r;
      }
    }
    __syncthreads();

    // ---- Step 2: grad = R @ Ks^T ; shrink + causal-mask alpha in-place ----
    const int n_st  = (t0 + TT) >> 4;              // live 16-col s-tiles
    const int total = 2 * n_st;
    for (int idx = wave; idx < total; idx += 8) {
      const int tr2 = idx / n_st;
      const int sc  = idx - tr2 * n_st;
      const int s0  = sc * 16;
      const int trow0 = t0 + tr2 * 16;
      if (s0 > trow0 + 15) continue;               // fully above diagonal
      const int ar = tr2 * 16 + mrow;
      f32x8 g0 = {}, g1 = {};
      {
        bf16x8 alo = *(const bf16x8*)(sR + ar * RP + bko);
        bf16x8 ahi = *(const bf16x8*)(sR + ar * RP + bko + 8);
        bf16x8 blo = *(const bf16x8*)(sKs + (s0 + mrow) * KP + bko);
        bf16x8 bhi = *(const bf16x8*)(sKs + (s0 + mrow) * KP + bko + 8);
        g0 = wmma_bf16(cat8(alo, ahi), cat8(blo, bhi), g0);
      }
      {
        bf16x8 alo = *(const bf16x8*)(sR + ar * RP + 32 + bko);
        bf16x8 ahi = *(const bf16x8*)(sR + ar * RP + 40 + bko);
        bf16x8 blo = *(const bf16x8*)(sKs + (s0 + mrow) * KP + 32 + bko);
        bf16x8 bhi = *(const bf16x8*)(sKs + (s0 + mrow) * KP + 40 + bko);
        g1 = wmma_bf16(cat8(alo, ahi), cat8(blo, bhi), g1);
      }
      const int sg = s0 + mrow;                    // global s column
      #pragma unroll
      for (int i = 0; i < 8; ++i) {
        const int ml = tr2 * 16 + i + ((lane < 16) ? 0 : 8); // row in t-tile
        const int tg = t0 + ml;
        const float e  = sEta[ml];
        const float th = e * SPARS;
        const float aold = (float)sAlpha[ml * SP + sg];
        const float z  = aold - e * (g0[i] + g1[i]);
        const float az = fabsf(z) - th;
        float anew = (az > 0.f) ? copysignf(az, z) : 0.f;
        if (sg > tg) anew = 0.f;
        sAlpha[ml * SP + sg] = (bf16_t)anew;
      }
    }
    __syncthreads();
  }

  // ---- Output: out = alpha @ V (same tiling as step 1, B from V^T) ----
  const bf16_t* gVT = VT + (size_t)b * DK * SS;
  f32x8 acc0 = {}, acc1 = {};
  const int nch = (t0 + TT + 31) >> 5;
  int kc = 0;
  for (; kc + 1 < nch; kc += 2) {
    {
      const int kb = kc * 32;
      __builtin_prefetch(gVT + (size_t)bn1 * SS + kb + 128, 0, 3);
      bf16x8 alo = *(const bf16x8*)(sAlpha + arow * SP + kb + bko);
      bf16x8 ahi = *(const bf16x8*)(sAlpha + arow * SP + kb + bko + 8);
      bf16x8 blo = *(const bf16x8*)(gVT + (size_t)bn1 * SS + kb + bko);
      bf16x8 bhi = *(const bf16x8*)(gVT + (size_t)bn1 * SS + kb + bko + 8);
      acc0 = wmma_bf16(cat8(alo, ahi), cat8(blo, bhi), acc0);
    }
    {
      const int kb = kc * 32 + 32;
      bf16x8 alo = *(const bf16x8*)(sAlpha + arow * SP + kb + bko);
      bf16x8 ahi = *(const bf16x8*)(sAlpha + arow * SP + kb + bko + 8);
      bf16x8 blo = *(const bf16x8*)(gVT + (size_t)bn1 * SS + kb + bko);
      bf16x8 bhi = *(const bf16x8*)(gVT + (size_t)bn1 * SS + kb + bko + 8);
      acc1 = wmma_bf16(cat8(alo, ahi), cat8(blo, bhi), acc1);
    }
  }
  if (kc < nch) {
    const int kb = kc * 32;
    bf16x8 alo = *(const bf16x8*)(sAlpha + arow * SP + kb + bko);
    bf16x8 ahi = *(const bf16x8*)(sAlpha + arow * SP + kb + bko + 8);
    bf16x8 blo = *(const bf16x8*)(gVT + (size_t)bn1 * SS + kb + bko);
    bf16x8 bhi = *(const bf16x8*)(gVT + (size_t)bn1 * SS + kb + bko + 8);
    acc0 = wmma_bf16(cat8(alo, ahi), cat8(blo, bhi), acc0);
  }
  float* gO = out + ((size_t)b * SS + t0) * DK;
  #pragma unroll
  for (int i = 0; i < 8; ++i) {
    const int m = tr * 16 + i + ((lane < 16) ? 0 : 8);
    const int n = nc * 16 + mrow;
    gO[m * DK + n] = acc0[i] + acc1[i];
  }
}

// ---------------------------------------------------------------------------
extern "C" void kernel_launch(void* const* d_in, const int* in_sizes, int n_in,
                              void* d_out, int out_size, void* d_ws, size_t ws_size,
                              hipStream_t stream) {
  const float* x  = (const float*)d_in[0];
  const float* Wq = (const float*)d_in[1];
  const float* bq = (const float*)d_in[2];
  const float* Wk = (const float*)d_in[3];
  const float* bk = (const float*)d_in[4];
  const float* Wv = (const float*)d_in[5];
  const float* bv = (const float*)d_in[6];
  float* out = (float*)d_out;

  char* ws = (char*)d_ws;
  const size_t MB = 1024ull * 1024ull;
  float*  Qs  = (float*) (ws + 0);        // B*S*64 f32 = 2 MB
  float*  Ksf = (float*) (ws + 2 * MB);   // 2 MB
  bf16_t* Ksb = (bf16_t*)(ws + 4 * MB);   // 1 MB
  bf16_t* KsT = (bf16_t*)(ws + 5 * MB);   // 1 MB
  bf16_t* VT  = (bf16_t*)(ws + 6 * MB);   // 1 MB
  float*  eta = (float*) (ws + 7 * MB);   // 32 KB

  proj_kernel<<<(BB * SS) / 4, 256, 0, stream>>>(x, Wq, bq, Wk, bk, Wv, bv,
                                                 Qs, Ksf, Ksb, KsT, VT);
  eta_kernel<<<BB * SS, 256, 0, stream>>>(Ksf, eta);
  ista_kernel<<<BB * (SS / TT), 256, SMEM_BYTES, stream>>>(Qs, Ksb, KsT, VT, eta, out);
}
